// MultiHeadAttention_73589969649754
// MI455X (gfx1250) — matrix operand order repaired and checked by reading, not compiled, not run on hardware
//
#include <hip/hip_runtime.h>
#include <hip/hip_fp16.h>

typedef __attribute__((ext_vector_type(16))) _Float16 v16h;
typedef __attribute__((ext_vector_type(8)))  _Float16 v8h;
typedef __attribute__((ext_vector_type(8)))  float    v8f;

#define C_DIM 512
#define INV_TEMP 0.044194173824159216f   // 1/sqrt(512)
#define NEG_SLOPE 0.01f
#define LDSW 520   // padded LDS row stride in halves: 260 dwords == 4 (mod 64)

// ---------------------------------------------------------------------------
// f32 -> f16 elementwise conversion, 8 elements per thread.
// ---------------------------------------------------------------------------
__global__ void __launch_bounds__(256)
cvt_f32_to_f16(const float* __restrict__ src, _Float16* __restrict__ dst,
               size_t n) {
  size_t i = ((size_t)blockIdx.x * blockDim.x + threadIdx.x) * 8;
  if (i >= n) return;
  float4 a = *(const float4*)(src + i);
  float4 b = *(const float4*)(src + i + 4);
  v8h o;
  o[0] = (_Float16)a.x; o[1] = (_Float16)a.y;
  o[2] = (_Float16)a.z; o[3] = (_Float16)a.w;
  o[4] = (_Float16)b.x; o[5] = (_Float16)b.y;
  o[6] = (_Float16)b.z; o[7] = (_Float16)b.w;
  *(v8h*)(dst + i) = o;
}

// ---------------------------------------------------------------------------
// WMMA fragment k order used here, same for A and B (CDNA5 16x16x32 f16, wave32)
// A (16x32, MxK): lane l -> row M = l&15, group g = l>>4.
//   halves 0..15 = K (16g + 0..15)
// B (32x16, KxN): lane l -> col N = l&15, group g = l>>4.
//   halves 0..15 = K (16g + 0..15)  (contiguous -> one 32B access)
// ---------------------------------------------------------------------------
union F16x16 {
  v16h v;
  struct { v8h lo; v8h hi; } p;
};

__device__ __forceinline__ v16h load_a_frag(const _Float16* __restrict__ rowp,
                                            int k0, int g) {
  F16x16 u;
  u.v = *(const v16h*)(rowp + k0 + 16 * g);
  return u.v;
}

// ---------------------------------------------------------------------------
// GEMM: Out[m, n] = act( sum_k A[m,k] * B[n,k] (+ bias[n]) )
// A: M x 512 f16 row-major (global), B: Nout x 512 f16 row-major.
// Block = 8 waves sharing one 64-wide B strip staged in LDS (65KB of the
// WGP's 320KB). Each wave computes a 16x64 strip: A fragment (2x b128,
// software-pipelined from global) is reused across 4 WMMA tiles whose B
// fragments come from LDS (ds_load_b128 pairs, bank-conflict-free via the
// 520-half row pitch). grid = (ceil(M/128), Nout/64).
// ---------------------------------------------------------------------------
__global__ void __launch_bounds__(256)
gat_wmma_gemm_nt(const _Float16* __restrict__ A, const _Float16* __restrict__ B,
                 const float* __restrict__ bias, float* __restrict__ Out,
                 int M, int Nout, int leaky) {
  __shared__ _Float16 ldsB[64 * LDSW];

  const int lane = threadIdx.x & 31;
  const int wave = threadIdx.x >> 5;
  const int g    = lane >> 4;
  const int l15  = lane & 15;

  // ---- cooperative stage of the 64 x 512 B strip into LDS ---------------
  {
    const _Float16* bsrc = B + (size_t)(blockIdx.y * 64) * C_DIM;
    const int row = threadIdx.x >> 2;       // 0..63
    const int q   = threadIdx.x & 3;        // quarter of the row
#pragma unroll
    for (int i = 0; i < 16; ++i) {
      const int off = q * 128 + i * 8;
      *(v8h*)(&ldsB[row * LDSW + off]) =
          *(const v8h*)(bsrc + (size_t)row * C_DIM + off);
    }
  }
  __syncthreads();

  const int tmtile = blockIdx.x * 8 + wave;             // wave-uniform
  if (tmtile >= (M >> 4)) return;                       // whole wave exits

  const _Float16* arow = A + (size_t)(tmtile * 16 + l15) * C_DIM;
  const _Float16* br0 = &ldsB[(0 * 16 + l15) * LDSW];
  const _Float16* br1 = &ldsB[(1 * 16 + l15) * LDSW];
  const _Float16* br2 = &ldsB[(2 * 16 + l15) * LDSW];
  const _Float16* br3 = &ldsB[(3 * 16 + l15) * LDSW];

  v8f acc0 = {}, acc1 = {}, acc2 = {}, acc3 = {};

  v16h af = load_a_frag(arow, 0, g);
  int kb = 0;
  for (int k0 = 32; k0 < C_DIM; k0 += 32) {
    v16h afn = load_a_frag(arow, k0, g);    // overlap next A with WMMAs
    v16h b0 = *(const v16h*)(br0 + kb + 16 * g);
    v16h b1 = *(const v16h*)(br1 + kb + 16 * g);
    v16h b2 = *(const v16h*)(br2 + kb + 16 * g);
    v16h b3 = *(const v16h*)(br3 + kb + 16 * g);
    acc0 = __builtin_amdgcn_wmma_f32_16x16x32_f16(false, af, false, b0,
                                                  (short)0, acc0, false, false);
    acc1 = __builtin_amdgcn_wmma_f32_16x16x32_f16(false, af, false, b1,
                                                  (short)0, acc1, false, false);
    acc2 = __builtin_amdgcn_wmma_f32_16x16x32_f16(false, af, false, b2,
                                                  (short)0, acc2, false, false);
    acc3 = __builtin_amdgcn_wmma_f32_16x16x32_f16(false, af, false, b3,
                                                  (short)0, acc3, false, false);
    af = afn;
    kb = k0;
  }
  {
    v16h b0 = *(const v16h*)(br0 + kb + 16 * g);
    v16h b1 = *(const v16h*)(br1 + kb + 16 * g);
    v16h b2 = *(const v16h*)(br2 + kb + 16 * g);
    v16h b3 = *(const v16h*)(br3 + kb + 16 * g);
    acc0 = __builtin_amdgcn_wmma_f32_16x16x32_f16(false, af, false, b0,
                                                  (short)0, acc0, false, false);
    acc1 = __builtin_amdgcn_wmma_f32_16x16x32_f16(false, af, false, b1,
                                                  (short)0, acc1, false, false);
    acc2 = __builtin_amdgcn_wmma_f32_16x16x32_f16(false, af, false, b2,
                                                  (short)0, acc2, false, false);
    acc3 = __builtin_amdgcn_wmma_f32_16x16x32_f16(false, af, false, b3,
                                                  (short)0, acc3, false, false);
  }

  // C/D layout: VGPR r -> row M = r + 8*g, col N = lane & 15.
  v8f accs[4] = {acc0, acc1, acc2, acc3};
#pragma unroll
  for (int t = 0; t < 4; ++t) {
    const int col = blockIdx.y * 64 + t * 16 + l15;
    const float bb = bias ? bias[col] : 0.0f;
#pragma unroll
    for (int r = 0; r < 8; ++r) {
      const int row = tmtile * 16 + r + 8 * g;
      float h = accs[t][r] + bb;
      if (leaky) h = (h >= 0.0f) ? h : NEG_SLOPE * h;
      Out[(size_t)row * Nout + col] = h;
    }
  }
}

// ---------------------------------------------------------------------------
// Per-edge score: s[e] = dot(Kl[segL[e]], Kr[segR[e]]) / sqrt(C). Wave/edge.
// ---------------------------------------------------------------------------
__global__ void __launch_bounds__(256)
edge_scores(const float* __restrict__ Kl, const float* __restrict__ Kr,
            const int* __restrict__ segL, const int* __restrict__ segR,
            float* __restrict__ s, int E) {
  const int e = (int)((blockIdx.x * (long)blockDim.x + threadIdx.x) >> 5);
  const int lane = threadIdx.x & 31;
  if (e >= E) return;
  const int il = segL[e], ir = segR[e];
  const float4* a = (const float4*)(Kl + (size_t)il * C_DIM) + lane * 4;
  const float4* b = (const float4*)(Kr + (size_t)ir * C_DIM) + lane * 4;
  float acc = 0.0f;
#pragma unroll
  for (int j = 0; j < 4; ++j) {
    float4 x = a[j], y = b[j];
    acc += x.x * y.x + x.y * y.y + x.z * y.z + x.w * y.w;
  }
#pragma unroll
  for (int off = 16; off > 0; off >>= 1) acc += __shfl_xor(acc, off, 32);
  if (lane == 0) s[e] = acc * INV_TEMP;
}

// Monotone f32 <-> u32 mapping for atomic float max (init value 0u < any enc).
__device__ __forceinline__ unsigned enc_f32(float x) {
  unsigned b = __float_as_uint(x);
  return (b & 0x80000000u) ? ~b : (b | 0x80000000u);
}
__device__ __forceinline__ float dec_f32(unsigned u) {
  unsigned b = (u & 0x80000000u) ? (u & 0x7FFFFFFFu) : ~u;
  return __uint_as_float(b);
}

__global__ void __launch_bounds__(256)
edge_segmax(const float* __restrict__ s, const int* __restrict__ segL,
            const int* __restrict__ segR, unsigned* __restrict__ mLu,
            unsigned* __restrict__ mRu, int E) {
  const int e = blockIdx.x * blockDim.x + threadIdx.x;
  if (e >= E) return;
  const unsigned enc = enc_f32(s[e]);
  atomicMax(&mLu[segL[e]], enc);
  atomicMax(&mRu[segR[e]], enc);
}

__global__ void __launch_bounds__(256)
edge_expsum(const float* __restrict__ s, const int* __restrict__ segL,
            const int* __restrict__ segR, const unsigned* __restrict__ mLu,
            const unsigned* __restrict__ mRu, float* __restrict__ eL,
            float* __restrict__ eR, float* __restrict__ zL,
            float* __restrict__ zR, int E) {
  const int e = blockIdx.x * blockDim.x + threadIdx.x;
  if (e >= E) return;
  const float sv = s[e];
  const int il = segL[e], ir = segR[e];
  const float el = __expf(sv - dec_f32(mLu[il]));
  const float er = __expf(sv - dec_f32(mRu[ir]));
  eL[e] = el;
  eR[e] = er;
  atomicAdd(&zL[il], el);
  atomicAdd(&zR[ir], er);
}

// ---------------------------------------------------------------------------
// Scatter: msgL[segL[e]] += (eL/zL) * Vr[segR[e]];
//          msgR[segR[e]] += (eR/zR) * Vl[segL[e]].  Wave per edge.
// ---------------------------------------------------------------------------
__global__ void __launch_bounds__(256)
edge_scatter(const float* __restrict__ Vl, const float* __restrict__ Vr,
             const int* __restrict__ segL, const int* __restrict__ segR,
             const float* __restrict__ eL, const float* __restrict__ eR,
             const float* __restrict__ zL, const float* __restrict__ zR,
             float* __restrict__ msgL, float* __restrict__ msgR, int E) {
  const int e = (int)((blockIdx.x * (long)blockDim.x + threadIdx.x) >> 5);
  const int lane = threadIdx.x & 31;
  if (e >= E) return;
  const int il = segL[e], ir = segR[e];
  const float aL = eL[e] / zL[il];
  const float aR = eR[e] / zR[ir];
  const float4* vr = (const float4*)(Vr + (size_t)ir * C_DIM) + lane * 4;
  const float4* vl = (const float4*)(Vl + (size_t)il * C_DIM) + lane * 4;
  float* ml = msgL + (size_t)il * C_DIM + lane * 16;
  float* mr = msgR + (size_t)ir * C_DIM + lane * 16;
#pragma unroll
  for (int j = 0; j < 4; ++j) {
    float4 x = vr[j];
    atomicAdd(ml + 4 * j + 0, aL * x.x);
    atomicAdd(ml + 4 * j + 1, aL * x.y);
    atomicAdd(ml + 4 * j + 2, aL * x.z);
    atomicAdd(ml + 4 * j + 3, aL * x.w);
    float4 y = vl[j];
    atomicAdd(mr + 4 * j + 0, aR * y.x);
    atomicAdd(mr + 4 * j + 1, aR * y.y);
    atomicAdd(mr + 4 * j + 2, aR * y.z);
    atomicAdd(mr + 4 * j + 3, aR * y.w);
  }
}

// ---------------------------------------------------------------------------
extern "C" void kernel_launch(void* const* d_in, const int* in_sizes, int n_in,
                              void* d_out, int out_size, void* d_ws,
                              size_t ws_size, hipStream_t stream) {
  const float* node_left  = (const float*)d_in[0];
  const int*   segL       = (const int*)d_in[1];
  const float* node_right = (const float*)d_in[3];
  const int*   segR       = (const int*)d_in[4];
  const float* Wk = (const float*)d_in[6];
  const float* Wv = (const float*)d_in[7];
  const float* Wo = (const float*)d_in[8];
  const float* bo = (const float*)d_in[9];
  float* out = (float*)d_out;

  const int N = in_sizes[0] / C_DIM;
  const int E = in_sizes[1];
  const size_t NC = (size_t)N * C_DIM;
  const size_t CC = (size_t)C_DIM * C_DIM;

  // ---- workspace layout: f32 region first, then f16 region --------------
  float*    Kl   = (float*)d_ws;
  float*    Kr   = Kl + NC;
  float*    Vl   = Kr + NC;
  float*    Vr   = Vl + NC;
  float*    s    = Vr + NC;
  float*    eL   = s + E;
  float*    eR   = eL + E;
  unsigned* mLu  = (unsigned*)(eR + E);
  unsigned* mRu  = mLu + N;
  float*    zL   = (float*)(mRu + N);
  float*    zR   = zL + N;
  float*    msgL = zR + N;
  float*    msgR = msgL + NC;

  _Float16* nlH   = (_Float16*)(msgR + NC);
  _Float16* nrH   = nlH + NC;
  _Float16* WkH   = nrH + NC;
  _Float16* WvH   = WkH + CC;
  _Float16* WoH   = WvH + CC;
  _Float16* msgLH = WoH + CC;
  _Float16* msgRH = msgLH + NC;

  // zero m (enc-min), z, msg in one contiguous memset
  hipMemsetAsync(mLu, 0, ((size_t)4 * N + 2 * NC) * sizeof(float), stream);

  const dim3 blk(256);

  // ---- f32 -> f16 operand conversion ------------------------------------
  const int cblocks_nc = (int)((NC / 8 + 255) / 256);
  const int cblocks_cc = (int)((CC / 8 + 255) / 256);
  cvt_f32_to_f16<<<cblocks_nc, blk, 0, stream>>>(node_left,  nlH, NC);
  cvt_f32_to_f16<<<cblocks_nc, blk, 0, stream>>>(node_right, nrH, NC);
  cvt_f32_to_f16<<<cblocks_cc, blk, 0, stream>>>(Wk, WkH, CC);
  cvt_f32_to_f16<<<cblocks_cc, blk, 0, stream>>>(Wv, WvH, CC);
  cvt_f32_to_f16<<<cblocks_cc, blk, 0, stream>>>(Wo, WoH, CC);

  // ---- projection GEMMs: grid = (ceil(M/128), Nout/64) ------------------
  const dim3 ggrid((N / 16 + 7) / 8, C_DIM / 64);
  gat_wmma_gemm_nt<<<ggrid, blk, 0, stream>>>(nlH, WkH, nullptr, Kl, N, C_DIM, 0);
  gat_wmma_gemm_nt<<<ggrid, blk, 0, stream>>>(nrH, WkH, nullptr, Kr, N, C_DIM, 0);
  gat_wmma_gemm_nt<<<ggrid, blk, 0, stream>>>(nlH, WvH, nullptr, Vl, N, C_DIM, 0);
  gat_wmma_gemm_nt<<<ggrid, blk, 0, stream>>>(nrH, WvH, nullptr, Vr, N, C_DIM, 0);

  // ---- edge pipeline -----------------------------------------------------
  const int wave_blocks = (E + 7) / 8;        // one wave per edge, 8 waves/blk
  const int thr_blocks  = (E + 255) / 256;
  edge_scores<<<wave_blocks, blk, 0, stream>>>(Kl, Kr, segL, segR, s, E);
  edge_segmax<<<thr_blocks, blk, 0, stream>>>(s, segL, segR, mLu, mRu, E);
  edge_expsum<<<thr_blocks, blk, 0, stream>>>(s, segL, segR, mLu, mRu,
                                              eL, eR, zL, zR, E);
  edge_scatter<<<wave_blocks, blk, 0, stream>>>(Vl, Vr, segL, segR, eL, eR,
                                                zL, zR, msgL, msgR, E);

  // ---- output GEMMs with bias + LeakyReLU, straight into d_out ----------
  cvt_f32_to_f16<<<cblocks_nc, blk, 0, stream>>>(msgL, msgLH, NC);
  cvt_f32_to_f16<<<cblocks_nc, blk, 0, stream>>>(msgR, msgRH, NC);
  gat_wmma_gemm_nt<<<ggrid, blk, 0, stream>>>(msgLH, WoH, bo, out,      N, C_DIM, 1);
  gat_wmma_gemm_nt<<<ggrid, blk, 0, stream>>>(msgRH, WoH, bo, out + NC, N, C_DIM, 1);
}
